// SubsetGFlowNetTB_91001767068254
// MI455X (gfx1250) — compile-verified, matrix operand order checked by reading, not run on hardware
//
#include <hip/hip_runtime.h>
#include <stdint.h>

typedef __attribute__((ext_vector_type(16))) __bf16 v16bf;
typedef __attribute__((ext_vector_type(8)))  __bf16 v8bf;
typedef __attribute__((ext_vector_type(8)))  float  v8f;

#define KMAXV 16
#define NEG_INF_V (-1000000000.0f)

// ---------------------------------------------------------------------------
// Pack a row-major f32 weight matrix W[K=256][ncols] into bf16 tiles laid out
// exactly as the WMMA B-operand (16-bit B 32x16) wants them in VGPRs:
//   tile (kc, nt): 256 dwords; dword index = lane*8 + v
//   lane -> n = nt*16 + (lane&15);  half = lane>>4 -> kb = half*8
//   dword v packs K = koff, koff+1 where
//   koff = (v<4) ? kb+2v : kb+16+2(v-4)
// so the main kernel loads two contiguous b128 per lane per tile.
// ---------------------------------------------------------------------------
__global__ void pack_wmma_b(const float* __restrict__ W, uint32_t* __restrict__ out,
                            int ncols, int ntn) {
  int idx = blockIdx.x * 256 + threadIdx.x;
  int total = 8 * ntn * 256;
  if (idx >= total) return;
  int tile   = idx >> 8;
  int within = idx & 255;
  int lane = within >> 3;
  int v    = within & 7;
  int kc = tile / ntn;
  int nt = tile - kc * ntn;
  int l15 = lane & 15;
  int hh  = lane >> 4;
  int kb  = hh * 8;
  int koff = (v < 4) ? (kb + 2 * v) : (kb + 16 + 2 * (v - 4));
  int k0 = kc * 32 + koff;
  int n  = nt * 16 + l15;
  float f0 = (n < ncols) ? W[(size_t)k0 * ncols + n] : 0.0f;
  float f1 = (n < ncols) ? W[(size_t)(k0 + 1) * ncols + n] : 0.0f;
  __bf16 h0 = (__bf16)f0;
  __bf16 h1 = (__bf16)f1;
  uint32_t u0 = (uint32_t)__builtin_bit_cast(unsigned short, h0);
  uint32_t u1 = (uint32_t)__builtin_bit_cast(unsigned short, h1);
  out[idx] = u0 | (u1 << 16);
}

// ---------------------------------------------------------------------------
// Main kernel: 16 batch rows per workgroup, 256 threads (8 wave32).
// Entire 17-step recurrence stays on-chip.
// ---------------------------------------------------------------------------
__global__ void __launch_bounds__(256, 1)
gfn_traj_kernel(const int* __restrict__ actions,   // (B,17)
                const float* __restrict__ rp,      // (B,1)
                const float* __restrict__ W1,      // (515,256) f32
                const float* __restrict__ b1,      // (256)
                const float* __restrict__ b2,      // (256)
                const float* __restrict__ bh,      // (513)
                const uint32_t* __restrict__ w2pk, // packed bf16 W2
                const uint32_t* __restrict__ whpk, // packed bf16 Wh (pad 528)
                float* __restrict__ out, int B) {
  __shared__ __align__(16) __bf16 sh_h1[16 * 256];
  __shared__ __align__(16) __bf16 sh_h2[16 * 256];
  __shared__ float    sh_lg[16 * 528];
  __shared__ unsigned sh_sel[16][16];
  __shared__ float    sh_red[16][16];
  __shared__ float    sh_rmax[16];
  __shared__ int      sh_k[16];
  __shared__ float    sh_lp[16];
  __shared__ float    sh_rp[16];
  __shared__ int      sh_add[16];

  const int tid   = threadIdx.x;
  const int gRow0 = blockIdx.x * 16;
  const int wid   = tid >> 5;
  const int lane  = tid & 31;
  const int l15   = lane & 15;
  const int hh    = lane >> 4;
  const int kb    = hh * 8;      // also the D-matrix row offset (0 or 8)

  // per-thread slice of incremental W1 gather-sum: row rA, cols cA..cA+15
  const int rA = tid >> 4;
  const int cA = (tid & 15) * 16;
  float acc1[16];
#pragma unroll
  for (int i = 0; i < 16; ++i) acc1[i] = 0.0f;

  if (tid < 16) {
    sh_k[tid]  = 0;
    sh_lp[tid] = 0.0f;
    sh_rp[tid] = rp[gRow0 + tid];
    sh_add[tid] = -1;
  }
  sh_sel[tid >> 4][tid & 15] = 0u;
  __syncthreads();

  for (int step = 0; step < 17; ++step) {
    // ---- Phase 1: h1 = relu(x @ W1 + b1), incremental sparse form --------
    {
      float kf = (float)sh_k[rA];
      float sf = (float)step;
      float rv = sh_rp[rA];
#pragma unroll
      for (int i = 0; i < 16; ++i) {
        int c = cA + i;
        float v = acc1[i]
                + kf * W1[512 * 256 + c]
                + sf * W1[513 * 256 + c]
                + rv * W1[514 * 256 + c]
                + b1[c];
        sh_h1[rA * 256 + c] = (__bf16)fmaxf(v, 0.0f);
      }
    }
    __syncthreads();

    // ---- Phase 2: h2 = relu(h1 @ W2 + b2) via bf16 WMMA ------------------
    // Keep the N-tile loop rolled: the compiler hoists the A tiles into
    // persistent VGPRs across iterations and pipelines the B streams without
    // spilling (unrolling this loop was shown to cause scratch spills).
#pragma unroll 1
    for (int nt = wid; nt < 16; nt += 8) {
      v8f acc = {};
#pragma unroll
      for (int kc = 0; kc < 8; ++kc) {
        const __bf16* ap = sh_h1 + l15 * 256 + kc * 32 + kb;
        v8bf alo = *(const v8bf*)ap;          // K = kb .. kb+7
        v8bf ahi = *(const v8bf*)(ap + 16);   // K = kb+16 .. kb+23
        v16bf A = __builtin_shufflevector(alo, ahi,
                    0,1,2,3,4,5,6,7,8,9,10,11,12,13,14,15);
        const uint32_t* bp = w2pk + (size_t)(kc * 16 + nt) * 256 + lane * 8;
        v8bf blo = *(const v8bf*)bp;
        v8bf bhi = *(const v8bf*)(bp + 4);
        v16bf Bv = __builtin_shufflevector(blo, bhi,
                    0,1,2,3,4,5,6,7,8,9,10,11,12,13,14,15);
        acc = __builtin_amdgcn_wmma_f32_16x16x32_bf16(
                false, A, false, Bv, (short)0, acc, false, false);
      }
      int n = nt * 16 + l15;
      float bias = b2[n];
#pragma unroll
      for (int r = 0; r < 8; ++r) {
        int m = r + kb;
        float v = fmaxf(acc[r] + bias, 0.0f);
        sh_h2[m * 256 + n] = (__bf16)v;
      }
    }
    __syncthreads();

    // ---- Phase 3: logits = h2 @ Wh + bh, masked --------------------------
#pragma unroll 1
    for (int nt = wid; nt < 33; nt += 8) {
      v8f acc = {};
#pragma unroll
      for (int kc = 0; kc < 8; ++kc) {
        const __bf16* ap = sh_h2 + l15 * 256 + kc * 32 + kb;
        v8bf alo = *(const v8bf*)ap;
        v8bf ahi = *(const v8bf*)(ap + 16);
        v16bf A = __builtin_shufflevector(alo, ahi,
                    0,1,2,3,4,5,6,7,8,9,10,11,12,13,14,15);
        const uint32_t* bp = whpk + (size_t)(kc * 33 + nt) * 256 + lane * 8;
        v8bf blo = *(const v8bf*)bp;
        v8bf bhi = *(const v8bf*)(bp + 4);
        v16bf Bv = __builtin_shufflevector(blo, bhi,
                    0,1,2,3,4,5,6,7,8,9,10,11,12,13,14,15);
        acc = __builtin_amdgcn_wmma_f32_16x16x32_bf16(
                false, A, false, Bv, (short)0, acc, false, false);
      }
      int j = nt * 16 + l15;
      float bias = (j < 513) ? bh[j] : 0.0f;
#pragma unroll
      for (int r = 0; r < 8; ++r) {
        int m = r + kb;
        float lv = NEG_INF_V;
        if (j < 512) {
          bool selb = (sh_sel[m][j >> 5] >> (j & 31)) & 1u;
          if (!selb && sh_k[m] < KMAXV) lv = acc[r] + bias;
        } else if (j == 512) {
          if (sh_k[m] > 0) lv = acc[r] + bias;
        }
        sh_lg[m * 528 + j] = lv;
      }
    }
    __syncthreads();

    // ---- Phase 4: masked log-softmax, gather chosen, state decision ------
    {
      int row = tid >> 4, t16 = tid & 15;
      float pm = -3.0e38f;
      for (int j = t16; j < 513; j += 16) pm = fmaxf(pm, sh_lg[row * 528 + j]);
      sh_red[row][t16] = pm;
    }
    __syncthreads();
    if (tid < 16) {
      float m = sh_red[tid][0];
#pragma unroll
      for (int i = 1; i < 16; ++i) m = fmaxf(m, sh_red[tid][i]);
      sh_rmax[tid] = m;
    }
    __syncthreads();
    {
      int row = tid >> 4, t16 = tid & 15;
      float m = sh_rmax[row];
      float ps = 0.0f;
      for (int j = t16; j < 513; j += 16) ps += __expf(sh_lg[row * 528 + j] - m);
      sh_red[row][t16] = ps;
    }
    __syncthreads();
    if (tid < 16) {
      int row = tid;
      float s = 0.0f;
#pragma unroll
      for (int i = 0; i < 16; ++i) s += sh_red[row][i];
      int a = actions[(size_t)(gRow0 + row) * 17 + step];
      bool alive = (a >= 0);
      int asafe = a < 0 ? 0 : (a > 512 ? 512 : a);
      float chosen = sh_lg[row * 528 + asafe];
      if (alive) sh_lp[row] += chosen - sh_rmax[row] - __logf(s);
      bool stop = (a == 512) && alive;
      int doadd = -1;
      if (alive && !stop) {
        unsigned w  = sh_sel[row][asafe >> 5];
        unsigned bt = 1u << (asafe & 31);
        if (!(w & bt)) {          // jnp.maximum semantics: no double count
          sh_sel[row][asafe >> 5] = w | bt;
          sh_k[row] = sh_k[row] + 1;
          doadd = asafe;
        }
      }
      sh_add[row] = doadd;
    }
    __syncthreads();

    // ---- Phase 5: acc1 += W1[a,:] for newly selected asset ---------------
    {
      int as = sh_add[rA];
      if (as >= 0) {
#pragma unroll
        for (int i = 0; i < 16; ++i) acc1[i] += W1[(size_t)as * 256 + cA + i];
      }
    }
    __syncthreads();
  }

  if (tid < 16) out[gRow0 + tid] = sh_lp[tid];
}

// ---------------------------------------------------------------------------
extern "C" void kernel_launch(void* const* d_in, const int* in_sizes, int n_in,
                              void* d_out, int out_size, void* d_ws, size_t ws_size,
                              hipStream_t stream) {
  const int*   actions = (const int*)d_in[0];
  const float* rp      = (const float*)d_in[1];
  const float* W1      = (const float*)d_in[2];
  const float* b1      = (const float*)d_in[3];
  const float* W2      = (const float*)d_in[4];
  const float* b2      = (const float*)d_in[5];
  const float* Wh      = (const float*)d_in[6];
  const float* bh      = (const float*)d_in[7];
  float* out = (float*)d_out;
  const int B = in_sizes[1];            // rp has B elements

  uint32_t* w2pk = (uint32_t*)d_ws;                                     // 128 KB
  uint32_t* whpk = (uint32_t*)((char*)d_ws + (size_t)8 * 16 * 256 * 4); // 264 KB

  {
    int tot = 8 * 16 * 256;
    pack_wmma_b<<<(tot + 255) / 256, 256, 0, stream>>>(W2, w2pk, 256, 16);
  }
  {
    int tot = 8 * 33 * 256;
    pack_wmma_b<<<(tot + 255) / 256, 256, 0, stream>>>(Wh, whpk, 513, 33);
  }
  gfn_traj_kernel<<<B / 16, 256, 0, stream>>>(actions, rp, W1, b1, b2, bh,
                                              w2pk, whpk, out, B);
}
